// Predictor_798863917290
// MI455X (gfx1250) — matrix operand order repaired and checked by reading, not compiled, not run on hardware
//
#include <hip/hip_runtime.h>
#include <math.h>

// ---------------- problem constants ----------------
#define VV 1024
#define DD 1024
#define JJ 1024
#define BB 64
#define UU 512
#define TT (UU + 1)          // 513 steps
#define MT (BB * TT)         // 32832 rows
#define G3 (3 * DD)          // 3072

// scan partitioning
#define CHUNK 32
#define NWG (DD / CHUNK)     // 32 workgroups in the persistent scan kernel
#define LROWS (3 * CHUNK)    // 96 weight rows per WG (r,z,n slices)

// ---------------- workspace layout (bytes, all 256-aligned) ----------------
#define OFF_BAR    ((size_t)0)
#define OFF_EMBED  ((size_t)256)
#define OFF_WIH    (OFF_EMBED + (size_t)VV * DD * 2)
#define OFF_WHH    (OFF_WIH   + (size_t)G3 * DD * 2)
#define OFF_LW     (OFF_WHH   + (size_t)G3 * DD * 2)
#define OFF_HINIT  (OFF_LW    + (size_t)JJ * DD * 2)
#define OFF_HF32   (OFF_HINIT + (size_t)BB * DD * 2)
#define OFF_GX     (OFF_HF32  + (size_t)2 * BB * DD * 4)
#define OFF_HS     (OFF_GX    + (size_t)MT * G3 * 2)
// total ~287 MB

typedef __attribute__((ext_vector_type(16))) __bf16 v16bf;
typedef __attribute__((ext_vector_type(8)))  float  v8f;

union ABFrag { uint4 u4[2]; v16bf v; };

static __device__ __forceinline__ unsigned short f2bf(float f) {
  unsigned x = __float_as_uint(f);
  unsigned r = x + 0x7fffu + ((x >> 16) & 1u);   // round-to-nearest-even
  return (unsigned short)(r >> 16);
}
static __device__ __forceinline__ float bf2f(unsigned short h) {
  return __uint_as_float(((unsigned)h) << 16);
}

// ---------------- f32 -> bf16 conversion ----------------
__global__ void k_cvt(const float* __restrict__ src, unsigned short* __restrict__ dst, int n) {
  int idx = blockIdx.x * blockDim.x + threadIdx.x;
  int stride = gridDim.x * blockDim.x;
  for (int i = idx; i < n; i += stride) dst[i] = f2bf(src[i]);
}

// ---------------- initial state + barrier init ----------------
__global__ void k_init(const float* __restrict__ h0,
                       unsigned short* __restrict__ hInitBf,
                       float* __restrict__ hF32,
                       unsigned* __restrict__ bar) {
  int idx = blockIdx.x * blockDim.x + threadIdx.x;
  if (idx == 0) { bar[0] = 0u; bar[1] = 0u; }
  if (idx < BB * DD) {
    float v = h0[idx & (DD - 1)];
    hF32[idx]    = v;                 // ping buffer 0 = h_{-1}
    hInitBf[idx] = f2bf(v);
  }
}

// ---------------- GEMM 1: gx = gather(embed) @ W_ih^T + b_ih  (bf16 out) ----
// 256 thr = 8 waves; wave tile 32Mx64N (2 A-tiles x 4 B-tiles = 8 WMMA/k-iter);
// block tile 32M x 512N.
__global__ __launch_bounds__(256) void k_gx(
    const int* __restrict__ y,
    const unsigned short* __restrict__ embedBf,
    const unsigned short* __restrict__ wihBf,
    const float* __restrict__ bias_ih,
    unsigned short* __restrict__ gxBf) {
  const int lane = threadIdx.x & 31, w = threadIdx.x >> 5;
  const int mBase = blockIdx.y * 32;
  const int nBase = blockIdx.x * 512 + w * 64;
  const int lm = lane & 15, lg = lane >> 4;

  // A row gather for both M-tiles: row -> (b,u) -> token -> embed row
  const unsigned short* aRow[2];
#pragma unroll
  for (int mi = 0; mi < 2; ++mi) {
    int row = mBase + mi * 16 + lm;
    int b = row / TT, u = row % TT;
    int tok = (u == 0) ? 0 : y[b * UU + (u - 1)];
    aRow[mi] = embedBf + (size_t)tok * DD;
  }

  const unsigned short* bRow[4];
#pragma unroll
  for (int nt = 0; nt < 4; ++nt)
    bRow[nt] = wihBf + (size_t)(nBase + nt * 16 + lm) * DD;

  v8f acc[2][4] = {};
  for (int kk = 0; kk < DD; kk += 32) {
    ABFrag a[2];
#pragma unroll
    for (int mi = 0; mi < 2; ++mi) {
      a[mi].u4[0] = *(const uint4*)(aRow[mi] + kk + 8 * lg);
      a[mi].u4[1] = *(const uint4*)(aRow[mi] + kk + 16 + 8 * lg);
    }
#pragma unroll
    for (int nt = 0; nt < 4; ++nt) {
      ABFrag bb;
      bb.u4[0] = *(const uint4*)(bRow[nt] + kk + 8 * lg);
      bb.u4[1] = *(const uint4*)(bRow[nt] + kk + 16 + 8 * lg);
#pragma unroll
      for (int mi = 0; mi < 2; ++mi)
        acc[mi][nt] = __builtin_amdgcn_wmma_f32_16x16x32_bf16(
            false, a[mi].v, false, bb.v, (short)0, acc[mi][nt], false, false);
    }
  }
#pragma unroll
  for (int nt = 0; nt < 4; ++nt) {
    int col = nBase + nt * 16 + lm;
    float bi = bias_ih[col];
#pragma unroll
    for (int mi = 0; mi < 2; ++mi)
#pragma unroll
      for (int i = 0; i < 8; ++i) {
        int orow = mBase + mi * 16 + i + 8 * lg;
        gxBf[(size_t)orow * G3 + col] = f2bf(acc[mi][nt][i] + bi);
      }
  }
}

// ---------------- persistent GRU scan kernel -------------------------------
// 32 WGs x 256 thr. WG g owns hidden cols [32g,32g+32); preloads 96x1024 bf16
// W_hh slice into LDS (192KB) once; 513 steps of 64x1024x96 WMMA GEMM + gates
// + grid barrier + device-scope cache invalidate.
__global__ __launch_bounds__(256) void k_scan(
    const unsigned short* __restrict__ whhBf,
    const float* __restrict__ bias_hh,
    const unsigned short* __restrict__ gxBf,
    const unsigned short* __restrict__ hInitBf,
    float* __restrict__ hF32,            // 2 x [64][1024] ping-pong
    unsigned short* __restrict__ hsBf,   // [513][64][1024]
    unsigned* __restrict__ bar) {
  extern __shared__ char smem[];
  unsigned short* wLds = (unsigned short*)smem;                       // [96][1024]
  float* ghLds = (float*)(smem + (size_t)LROWS * DD * 2);             // [64][96]

  const int tid = threadIdx.x, lane = tid & 31, w = tid >> 5;
  const int mt = w & 3;        // batch tile 16*mt..
  const int ng = w >> 2;       // n-tile group (3 tiles each)
  const int lm = lane & 15, lg = lane >> 4;
  const int wgBase = blockIdx.x * CHUNK;

  // preload W_hh slice: local row lr -> global row (lr/32)*1024 + wgBase + lr%32
  for (int idx = tid; idx < LROWS * (DD / 8); idx += blockDim.x) {
    int lr = idx / (DD / 8);
    int kc = (idx % (DD / 8)) * 8;
    int grow = (lr / CHUNK) * DD + wgBase + (lr % CHUNK);
    *(uint4*)(wLds + lr * DD + kc) = *(const uint4*)(whhBf + (size_t)grow * DD + kc);
  }

  // hoist loop-invariant gate-phase state (each thread owns 8 fixed (m,c) pairs)
  float bhr[8], bhz[8], bhn[8];
  int ghIdx[8], hIdx[8];
  size_t gxIdx[8];
#pragma unroll
  for (int k = 0; k < 8; ++k) {
    int e = tid + k * 256;
    int m = e >> 5, c = e & (CHUNK - 1);
    int col = wgBase + c;
    bhr[k] = bias_hh[col];
    bhz[k] = bias_hh[DD + col];
    bhn[k] = bias_hh[2 * DD + col];
    ghIdx[k] = m * LROWS + c;
    hIdx[k]  = m * DD + col;
    gxIdx[k] = (size_t)m * TT * G3 + col;
  }
  __syncthreads();

  for (int t = 0; t < TT; ++t) {
    const unsigned short* hSrc = (t == 0) ? hInitBf : (hsBf + (size_t)(t - 1) * (BB * DD));
    const unsigned short* aRow = hSrc + (size_t)(mt * 16 + lm) * DD;
    v8f acc[3] = {};
    for (int kk = 0; kk < DD; kk += 32) {
      ABFrag a;
      a.u4[0] = *(const uint4*)(aRow + kk + 8 * lg);
      a.u4[1] = *(const uint4*)(aRow + kk + 16 + 8 * lg);
#pragma unroll
      for (int j = 0; j < 3; ++j) {
        int lr = (ng * 3 + j) * 16 + lm;
        ABFrag bb;
        bb.u4[0] = *(const uint4*)(wLds + lr * DD + kk + 8 * lg);
        bb.u4[1] = *(const uint4*)(wLds + lr * DD + kk + 16 + 8 * lg);
        acc[j] = __builtin_amdgcn_wmma_f32_16x16x32_bf16(
            false, a.v, false, bb.v, (short)0, acc[j], false, false);
      }
    }
    // spill gh (pre-activations) to LDS
#pragma unroll
    for (int j = 0; j < 3; ++j) {
      int nloc = (ng * 3 + j) * 16 + lm;
#pragma unroll
      for (int i = 0; i < 8; ++i) {
        int m = mt * 16 + i + 8 * lg;
        ghLds[m * LROWS + nloc] = acc[j][i];
      }
    }
    __syncthreads();

    // gates: r,z,n for this WG's 64 batch rows x 32 owned cols
    const float* hPrev = hF32 + (size_t)(t & 1) * (BB * DD);
    float* hCur        = hF32 + (size_t)((t + 1) & 1) * (BB * DD);
    const size_t tOff = (size_t)t * G3;
#pragma unroll
    for (int k = 0; k < 8; ++k) {
      const unsigned short* gxp = gxBf + gxIdx[k] + tOff;
      float gr = bf2f(gxp[0]);
      float gz = bf2f(gxp[DD]);
      float gn = bf2f(gxp[2 * DD]);
      float ghr = ghLds[ghIdx[k]]             + bhr[k];
      float ghz = ghLds[ghIdx[k] + CHUNK]     + bhz[k];
      float ghn = ghLds[ghIdx[k] + 2 * CHUNK] + bhn[k];
      float r = 1.f / (1.f + __expf(-(gr + ghr)));
      float z = 1.f / (1.f + __expf(-(gz + ghz)));
      float n = tanhf(gn + r * ghn);
      float h = (1.f - z) * n + z * hPrev[hIdx[k]];
      hCur[hIdx[k]] = h;
      hsBf[(size_t)t * (BB * DD) + hIdx[k]] = f2bf(h);
    }

    // grid-wide barrier (all 32 WGs resident)
    __syncthreads();
    __threadfence();
    if (tid == 0) {
      unsigned gen = atomicAdd(&bar[1], 0u);
      unsigned prev = atomicAdd(&bar[0], 1u);
      if (prev == (unsigned)(gridDim.x - 1)) {
        atomicExch(&bar[0], 0u);
        atomicAdd(&bar[1], 1u);
      } else {
        while (atomicAdd(&bar[1], 0u) == gen) __builtin_amdgcn_s_sleep(8);
      }
    }
    __syncthreads();
    // make other WGs' hs/hF32 writes visible: invalidate near caches (DEV scope)
    asm volatile("global_inv scope:SCOPE_DEV\n\ts_wait_loadcnt 0x0" ::: "memory");
    __threadfence();
  }
}

// ---------------- GEMM 2: out = hs @ linear_w^T + linear_b (f32 out) -------
// wave tile 32Mx64N, block tile 32Mx512N.
__global__ __launch_bounds__(256) void k_out(
    const unsigned short* __restrict__ hsBf,
    const unsigned short* __restrict__ lwBf,
    const float* __restrict__ linear_b,
    float* __restrict__ out) {
  const int lane = threadIdx.x & 31, w = threadIdx.x >> 5;
  const int mBase = blockIdx.y * 32;
  const int nBase = blockIdx.x * 512 + w * 64;
  const int lm = lane & 15, lg = lane >> 4;

  const unsigned short* aRow[2];
#pragma unroll
  for (int mi = 0; mi < 2; ++mi) {
    int row = mBase + mi * 16 + lm;
    int b = row / TT, u = row % TT;
    aRow[mi] = hsBf + (size_t)(u * BB + b) * DD;
  }

  const unsigned short* bRow[4];
#pragma unroll
  for (int nt = 0; nt < 4; ++nt)
    bRow[nt] = lwBf + (size_t)(nBase + nt * 16 + lm) * DD;

  v8f acc[2][4] = {};
  for (int kk = 0; kk < DD; kk += 32) {
    ABFrag a[2];
#pragma unroll
    for (int mi = 0; mi < 2; ++mi) {
      a[mi].u4[0] = *(const uint4*)(aRow[mi] + kk + 8 * lg);
      a[mi].u4[1] = *(const uint4*)(aRow[mi] + kk + 16 + 8 * lg);
    }
#pragma unroll
    for (int nt = 0; nt < 4; ++nt) {
      ABFrag bb;
      bb.u4[0] = *(const uint4*)(bRow[nt] + kk + 8 * lg);
      bb.u4[1] = *(const uint4*)(bRow[nt] + kk + 16 + 8 * lg);
#pragma unroll
      for (int mi = 0; mi < 2; ++mi)
        acc[mi][nt] = __builtin_amdgcn_wmma_f32_16x16x32_bf16(
            false, a[mi].v, false, bb.v, (short)0, acc[mi][nt], false, false);
    }
  }
#pragma unroll
  for (int nt = 0; nt < 4; ++nt) {
    int col = nBase + nt * 16 + lm;
    float bi = linear_b[col];
#pragma unroll
    for (int mi = 0; mi < 2; ++mi)
#pragma unroll
      for (int i = 0; i < 8; ++i) {
        int orow = mBase + mi * 16 + i + 8 * lg;
        out[(size_t)orow * JJ + col] = acc[mi][nt][i] + bi;
      }
  }
}

// ---------------- launch ----------------------------------------------------
extern "C" void kernel_launch(void* const* d_in, const int* in_sizes, int n_in,
                              void* d_out, int out_size, void* d_ws, size_t ws_size,
                              hipStream_t stream) {
  const int*   y   = (const int*)  d_in[0];
  const float* emb = (const float*)d_in[1];
  const float* wih = (const float*)d_in[2];
  const float* bih = (const float*)d_in[3];
  const float* whh = (const float*)d_in[4];
  const float* bhh = (const float*)d_in[5];
  const float* lw  = (const float*)d_in[6];
  const float* lb  = (const float*)d_in[7];
  const float* h0  = (const float*)d_in[8];

  char* ws = (char*)d_ws;
  unsigned*       bar     = (unsigned*)      (ws + OFF_BAR);
  unsigned short* embedBf = (unsigned short*)(ws + OFF_EMBED);
  unsigned short* wihBf   = (unsigned short*)(ws + OFF_WIH);
  unsigned short* whhBf   = (unsigned short*)(ws + OFF_WHH);
  unsigned short* lwBf    = (unsigned short*)(ws + OFF_LW);
  unsigned short* hInitBf = (unsigned short*)(ws + OFF_HINIT);
  float*          hF32    = (float*)         (ws + OFF_HF32);
  unsigned short* gxBf    = (unsigned short*)(ws + OFF_GX);
  unsigned short* hsBf    = (unsigned short*)(ws + OFF_HS);

  k_cvt<<<256, 256, 0, stream>>>(emb, embedBf, VV * DD);
  k_cvt<<<256, 256, 0, stream>>>(wih, wihBf, G3 * DD);
  k_cvt<<<256, 256, 0, stream>>>(whh, whhBf, G3 * DD);
  k_cvt<<<256, 256, 0, stream>>>(lw,  lwBf,  JJ * DD);
  k_init<<<(BB * DD + 255) / 256, 256, 0, stream>>>(h0, hInitBf, hF32, bar);

  dim3 g1(G3 / 512, MT / 32);
  k_gx<<<g1, 256, 0, stream>>>(y, embedBf, wihBf, bih, gxBf);

  size_t smem = (size_t)LROWS * DD * 2 + (size_t)BB * LROWS * 4;  // 221184 B
  k_scan<<<NWG, 256, smem, stream>>>(whhBf, bhh, gxBf, hInitBf, hF32, hsBf, bar);

  dim3 g2(JJ / 512, MT / 32);
  k_out<<<g2, 256, 0, stream>>>(hsBf, lwBf, lb, (float*)d_out);
}
